// AttentionMultipleHead_44504451121595
// MI455X (gfx1250) — matrix operand order repaired and checked by reading, not compiled, not run on hardware
//
#include <hip/hip_runtime.h>
#include <hip/hip_bf16.h>

// Problem constants (fixed by the reference)
#define BB 4
#define SS 2048
#define DD 512
#define HH 8
#define EE 512
#define SCALEF 0.044194173824159216f  // 1/sqrt(512)

// Use CDNA5 Tensor Data Mover for LDS tile staging (CDNA5 ISA)
#define USE_TDM 1

typedef __bf16 bf16_t;
typedef unsigned int u32;
typedef __attribute__((ext_vector_type(16))) __bf16 v16bf;
typedef __attribute__((ext_vector_type(8)))  __bf16 v8bf;
typedef __attribute__((ext_vector_type(8)))  float  v8f;
typedef __attribute__((ext_vector_type(4)))  u32    v4u;
typedef __attribute__((ext_vector_type(8)))  u32    v8u;

#define WMMA_BF16(a, b, c) \
  __builtin_amdgcn_wmma_f32_16x16x32_bf16(false, (a), false, (b), (short)0, (c), false, false)

// D# group1 dword0 presets: data_size=2B (bits17:16=1), pad_enable (bit20),
// pad_interval (bits24:22, code k -> 2^(k+1) DWORDs), pad_amount (bits31:25, n-1 DWORDs).
// Pad = 4 DWORDs (8 bf16) after each tile row -> reproduces the padded LDS layouts below.
#define TDM_DW0_ROW16DW  0x06D10000u  // 32-elem rows  (16 DW) + 4 DW pad  -> stride 40
#define TDM_DW0_ROW64DW  0x07510000u  // 128-elem rows (64 DW) + 4 DW pad  -> stride 136
#define TDM_DW0_ROW256DW 0x07D10000u  // 512-elem rows (256 DW) + 4 DW pad -> stride 520

__device__ __forceinline__ int lane_id() { return (int)(threadIdx.x & 31); }

// Low 32 bits of a generic LDS pointer == LDS byte offset (ISA 10.2 aperture rules).
__device__ __forceinline__ u32 lds_off(const void* p) { return (u32)(uintptr_t)p; }

#if USE_TDM
// Issue TENSOR_LOAD_TO_LDS for a 2D tile: tile_x bf16 elems per row, tile_y rows,
// global row stride `stride_elems`. Tracked by TENSORcnt.
__device__ __forceinline__ void tdm_load_2d(u32 lds_byte, const void* gaddr,
                                            u32 tile_x, u32 tile_y, u32 stride_elems,
                                            u32 dw0) {
  unsigned long long ga = (unsigned long long)(uintptr_t)gaddr;
  v4u g0;
  g0[0] = 1u;                                            // count=1, no gather
  g0[1] = lds_byte;                                      // lds_addr
  g0[2] = (u32)ga;                                       // global_addr[31:0]
  g0[3] = (u32)((ga >> 32) & 0x1FFFFFFu) | (2u << 30);   // global_addr[56:32], type=2
  v8u g1;
  g1[0] = dw0;                                           // mask=0, data_size, pad cfg
  g1[1] = (tile_x & 0xFFFFu) << 16;                      // tensor_dim0[15:0] = tile_x
  g1[2] = (tile_x >> 16) | ((tile_y & 0xFFFFu) << 16);   // tensor_dim0 hi, tensor_dim1 lo
  g1[3] = (tile_y >> 16) | (tile_x << 16);               // tensor_dim1 hi, tile_dim0
  g1[4] = tile_y;                                        // tile_dim1 (tile_dim2 = 0)
  g1[5] = stride_elems;                                  // tensor_dim0_stride[31:0]
  g1[6] = 0u;
  g1[7] = 0u;
  asm volatile("tensor_load_to_lds %0, %1" :: "s"(g0), "s"(g1) : "memory");
}
#endif

// ---------------------------------------------------------------------------
// Fragment loaders, contiguous k order: element s of lane l holds k = 16*(l/16) + s (A and B)
// ---------------------------------------------------------------------------

// A fragment: 16x32 bf16, row-major source (global or LDS), row stride `ld`.
__device__ __forceinline__ v16bf afrag_mem(const bf16_t* base, int ld) {
  int lane = lane_id();
  const bf16_t* p = base + (size_t)(lane & 15) * ld + ((lane >> 4) << 4);
  v8bf lo = *(const v8bf*)p;
  v8bf hi = *(const v8bf*)(p + 8);
  v16bf a;
#pragma unroll
  for (int e = 0; e < 8; ++e) { a[e] = lo[e]; a[e + 8] = hi[e]; }
  return a;
}

// B fragment where source is K stored row-major [t, e]: B(kk,n) = K[t0+n][e0+kk].
__device__ __forceinline__ v16bf bfragKT_g(const bf16_t* kbase, int ld) {
  int lane = lane_id();
  const bf16_t* p = kbase + (size_t)(lane & 15) * ld + ((lane >> 4) << 4);
  v8bf lo = *(const v8bf*)p;
  v8bf hi = *(const v8bf*)(p + 8);
  v16bf b;
#pragma unroll
  for (int e = 0; e < 8; ++e) { b[e] = lo[e]; b[e + 8] = hi[e]; }
  return b;
}

// B fragment from a row-major [K x N] LDS tile (column gather).
__device__ __forceinline__ v16bf bfrag_lds(const bf16_t* base, int ld) {
  int lane = lane_id();
  const bf16_t* p = base + (size_t)((lane >> 4) << 4) * ld + (lane & 15);
  v16bf b;
#pragma unroll
  for (int e = 0; e < 16; ++e) b[e] = p[(size_t)e * ld];
  return b;
}

// ---------------------------------------------------------------------------
// fp32 -> bf16 convert
// ---------------------------------------------------------------------------
__global__ void cvt_bf16_k(const float* __restrict__ src, bf16_t* __restrict__ dst, int n) {
  int i = blockIdx.x * blockDim.x + threadIdx.x;
  int stride = gridDim.x * blockDim.x;
  for (; i < n; i += stride) dst[i] = (__bf16)src[i];
}

// ---------------------------------------------------------------------------
// 128x128 tiled WMMA GEMM, K-step 32, TDM double-buffered LDS staging.
// 8 waves as 2(M) x 4(N); each wave owns 4x2 16x16 accumulator tiles.
// MODE 0: C = X * W[h] + bq[h] -> bf16 [B,H,S,E];  MODE 1: C = O * Wp + bp -> f32
// ---------------------------------------------------------------------------
template <int MODE>
__global__ __launch_bounds__(256) void gemm128(const bf16_t* __restrict__ A, int lda,
                                               const bf16_t* __restrict__ Bm, int ldb, int K,
                                               const float* __restrict__ bias,
                                               void* __restrict__ Out) {
#if USE_TDM
  __shared__ __align__(16) bf16_t lA[2][128][40];
  __shared__ __align__(16) bf16_t lB[2][32][136];
#else
  __shared__ __align__(16) bf16_t lA[1][128][40];
  __shared__ __align__(16) bf16_t lB[1][32][136];
#endif

  const int tid  = threadIdx.x;
  const int lane = lane_id();
  const int wave = tid >> 5;
  const int m0 = (wave >> 2) * 64;
  const int n0 = (wave & 3) * 32;
  const int rowBase = blockIdx.y * 128;
  const int nBase   = blockIdx.x * 128;

  const bf16_t* Bm2 = Bm + (size_t)blockIdx.z * DD * EE;  // head offset (MODE 0)
  const bf16_t* Ab  = A + (size_t)rowBase * lda;
  const bf16_t* Bb  = Bm2 + nBase;
  const int nk = K / 32;

  v8f acc[4][2];
#pragma unroll
  for (int i = 0; i < 4; ++i)
#pragma unroll
    for (int j = 0; j < 2; ++j)
#pragma unroll
      for (int r = 0; r < 8; ++r) acc[i][j][r] = 0.0f;

#if USE_TDM
  if (wave == 0) {
    tdm_load_2d(lds_off(&lA[0][0][0]), Ab, 32, 128, (u32)lda, TDM_DW0_ROW16DW);
    tdm_load_2d(lds_off(&lB[0][0][0]), Bb, 128, 32, (u32)ldb, TDM_DW0_ROW64DW);
  }
#endif

  for (int kt = 0; kt < nk; ++kt) {
#if USE_TDM
    const int buf = kt & 1;
    if (wave == 0) {
      if (kt + 1 < nk) {  // prefetch next K-slab into the other buffer via TDM
        tdm_load_2d(lds_off(&lA[buf ^ 1][0][0]), Ab + (kt + 1) * 32, 32, 128,
                    (u32)lda, TDM_DW0_ROW16DW);
        tdm_load_2d(lds_off(&lB[buf ^ 1][0][0]), Bb + (size_t)(kt + 1) * 32 * ldb, 128, 32,
                    (u32)ldb, TDM_DW0_ROW64DW);
        __builtin_amdgcn_s_wait_tensorcnt(2);  // in-order: current buffer complete
      } else {
        __builtin_amdgcn_s_wait_tensorcnt(0);
      }
    }
#else
    const int buf = 0;
    const int k0 = kt * 32;
#pragma unroll
    for (int c = 0; c < 2; ++c) {
      int idx = (tid + c * 256) * 8;
      int row = idx >> 5, col = idx & 31;
      *(v8bf*)&lA[0][row][col] = *(const v8bf*)(A + (size_t)(rowBase + row) * lda + k0 + col);
    }
#pragma unroll
    for (int c = 0; c < 2; ++c) {
      int idx = (tid + c * 256) * 8;
      int kk = idx >> 7, n = idx & 127;
      *(v8bf*)&lB[0][kk][n] = *(const v8bf*)(Bm2 + (size_t)(k0 + kk) * ldb + nBase + n);
    }
#endif
    __syncthreads();

    v16bf af[4], bfv[2];
#pragma unroll
    for (int i = 0; i < 4; ++i) af[i] = afrag_mem(&lA[buf][m0 + i * 16][0], 40);
#pragma unroll
    for (int j = 0; j < 2; ++j) bfv[j] = bfrag_lds(&lB[buf][0][n0 + j * 16], 136);
#pragma unroll
    for (int i = 0; i < 4; ++i)
#pragma unroll
      for (int j = 0; j < 2; ++j) acc[i][j] = WMMA_BF16(af[i], bfv[j], acc[i][j]);
    __syncthreads();
  }

  // Epilogue: C slot r -> row = r + 8*(lane>=16), col = lane&15
#pragma unroll
  for (int i = 0; i < 4; ++i)
#pragma unroll
    for (int j = 0; j < 2; ++j)
#pragma unroll
      for (int r = 0; r < 8; ++r) {
        int grow = rowBase + m0 + i * 16 + r + ((lane >> 4) << 3);
        int gcol = nBase + n0 + j * 16 + (lane & 15);
        if (MODE == 0) {
          bf16_t* Op = (bf16_t*)Out;
          float v = acc[i][j][r] + bias[blockIdx.z * EE + gcol];
          int bi = grow >> 11;  // / S
          int s  = grow & (SS - 1);
          Op[(((size_t)bi * HH + blockIdx.z) * SS + s) * EE + gcol] = (__bf16)v;
        } else {
          float* Op = (float*)Out;
          Op[(size_t)grow * DD + gcol] = acc[i][j][r] + bias[gcol];
        }
      }
}

// ---------------------------------------------------------------------------
// Single-pass fused flash attention. Block = 16 query rows x full E = 512.
// Per 32-key tile, each of 8 waves computes a PARTIAL score tile over its own
// 64-wide slice of the E-contraction (4 WMMAs), partials are reduced across
// waves in LDS, every wave redundantly runs the (identical) online-softmax
// bookkeeping, then all waves do 4 PV WMMAs into their own 64-wide E chunk.
// => 8 WMMAs per wave per tile, perfectly balanced; scores computed once.
// V tiles stream in via TDM, overlapped with the score WMMAs.
// ---------------------------------------------------------------------------
__global__ __launch_bounds__(256) void attn_fused(const bf16_t* __restrict__ Q,
                                                  const bf16_t* __restrict__ Kc,
                                                  const bf16_t* __restrict__ V,
                                                  bf16_t* __restrict__ O) {
  __shared__ __align__(16) bf16_t Vl[32][520];     // V tile: 32 keys x 512
  __shared__ __align__(16) bf16_t Pl[16][40];      // P tile: 16 rows x 32 keys
  __shared__ float Sp[16][8][32];                  // per-(wave,half) partial C tiles
  __shared__ float Ssum[2][8][32];                 // reduced C tiles (2 key-halves)

  const int tid  = threadIdx.x;
  const int lane = lane_id();
  const int wave = tid >> 5;
  const int bh   = blockIdx.y;
  const int b    = bh >> 3;
  const int h    = bh & 7;
  const int srow = blockIdx.x * 16;
  const int hi8  = (lane >> 4) << 3;   // 0 or 8: row offset of this half-wave's C slots

  // This wave's 64-wide slice of the E-contraction: k-steps 2*wave, 2*wave+1.
  const bf16_t* Qbase = Q + ((size_t)bh * SS + srow) * EE;
  v16bf qf[2];
#pragma unroll
  for (int i = 0; i < 2; ++i) qf[i] = afrag_mem(Qbase + (2 * wave + i) * 32, EE);

  float m[8], l[8];
#pragma unroll
  for (int r = 0; r < 8; ++r) { m[r] = -3.0e38f; l[r] = 0.0f; }

  v8f oacc[4];
#pragma unroll
  for (int j = 0; j < 4; ++j)
#pragma unroll
    for (int r = 0; r < 8; ++r) oacc[j][r] = 0.0f;

  for (int t0 = 0; t0 < SS; t0 += 32) {
#if USE_TDM
    if (wave == 0)  // kick the V-tile DMA; it overlaps the score WMMAs below
      tdm_load_2d(lds_off(&Vl[0][0]), V + ((size_t)bh * SS + t0) * EE, 512, 32,
                  (u32)EE, TDM_DW0_ROW256DW);
#else
#pragma unroll
    for (int c = 0; c < 8; ++c) {
      int idx = (tid + c * 256) * 8;
      int key = idx >> 9, e = idx & 511;
      *(v8bf*)&Vl[key][e] = *(const v8bf*)(V + ((size_t)bh * SS + t0 + key) * EE + e);
    }
#endif
    // Partial scores over this wave's E-slice, for both 16-key halves.
#pragma unroll
    for (int half = 0; half < 2; ++half) {
      const bf16_t* Kb = Kc + ((size_t)bh * SS + t0 + half * 16) * EE;
      if (t0 + 32 < SS)
        __builtin_prefetch(Kb + (size_t)32 * EE + (2 * wave) * 32, 0, 1);
      v8f sacc;
#pragma unroll
      for (int r = 0; r < 8; ++r) sacc[r] = 0.0f;
#pragma unroll
      for (int i = 0; i < 2; ++i) {
        v16bf bk = bfragKT_g(Kb + (2 * wave + i) * 32, EE);
        sacc = WMMA_BF16(qf[i], bk, sacc);
      }
#pragma unroll
      for (int r = 0; r < 8; ++r) Sp[wave * 2 + half][r][lane] = sacc[r];
    }
    __syncthreads();

    // Cross-wave reduction of partial score tiles: 512 values, 2 per thread.
#pragma unroll
    for (int c = 0; c < 2; ++c) {
      int idx = tid + c * 256;
      int hf = idx >> 8, r = (idx >> 5) & 7, ln = idx & 31;
      float s = 0.0f;
#pragma unroll
      for (int w = 0; w < 8; ++w) s += Sp[w * 2 + hf][r][ln];
      Ssum[hf][r][ln] = s;
    }
    __syncthreads();

    // Online softmax bookkeeping (identical in every wave => fac shared for free).
    float fac[8];
#pragma unroll
    for (int r = 0; r < 8; ++r) {
      float s0 = Ssum[0][r][lane] * SCALEF;
      float s1 = Ssum[1][r][lane] * SCALEF;
      float t = fmaxf(s0, s1);
#pragma unroll
      for (int off = 1; off < 16; off <<= 1) t = fmaxf(t, __shfl_xor(t, off, 32));
      float nm = fmaxf(m[r], t);
      fac[r] = __expf(m[r] - nm);
      m[r] = nm;
      float p0 = __expf(s0 - nm);
      float p1 = __expf(s1 - nm);
      float ps = p0 + p1;
#pragma unroll
      for (int off = 1; off < 16; off <<= 1) ps += __shfl_xor(ps, off, 32);
      l[r] = l[r] * fac[r] + ps;
      if (wave == 0) {  // publish P (bf16) once
        Pl[r + hi8][lane & 15] = (__bf16)p0;
        Pl[r + hi8][16 + (lane & 15)] = (__bf16)p1;
      }
    }
#if USE_TDM
    if (wave == 0) __builtin_amdgcn_s_wait_tensorcnt(0);
#endif
    __syncthreads();

    // Rescale O and accumulate P*V over this wave's 64-wide E chunk.
#pragma unroll
    for (int j = 0; j < 4; ++j)
#pragma unroll
      for (int r = 0; r < 8; ++r) oacc[j][r] *= fac[r];
    v16bf ap = afrag_mem(&Pl[0][0], 40);
#pragma unroll
    for (int j = 0; j < 4; ++j) {
      v16bf bv = bfrag_lds(&Vl[0][wave * 64 + j * 16], 520);
      oacc[j] = WMMA_BF16(ap, bv, oacc[j]);
    }
    __syncthreads();
  }

  float linv[8];
#pragma unroll
  for (int r = 0; r < 8; ++r) linv[r] = 1.0f / l[r];

#pragma unroll
  for (int j = 0; j < 4; ++j)
#pragma unroll
    for (int r = 0; r < 8; ++r) {
      int s   = srow + r + hi8;
      int col = wave * 64 + j * 16 + (lane & 15);
      O[((size_t)b * SS + s) * (HH * EE) + (size_t)h * EE + col] = (__bf16)(oacc[j][r] * linv[r]);
    }
}

// ---------------------------------------------------------------------------
// Host launcher
// ---------------------------------------------------------------------------
extern "C" void kernel_launch(void* const* d_in, const int* in_sizes, int n_in,
                              void* d_out, int out_size, void* d_ws, size_t ws_size,
                              hipStream_t stream) {
  (void)in_sizes; (void)n_in; (void)out_size; (void)ws_size;

  const float* x  = (const float*)d_in[0];
  const float* Wq = (const float*)d_in[1];
  const float* bq = (const float*)d_in[2];
  const float* Wk = (const float*)d_in[3];
  const float* bk = (const float*)d_in[4];
  const float* Wv = (const float*)d_in[5];
  const float* bv = (const float*)d_in[6];
  const float* Wp = (const float*)d_in[7];
  const float* bp = (const float*)d_in[8];
  float* out = (float*)d_out;

  const size_t nX  = (size_t)BB * SS * DD;
  const size_t nW  = (size_t)HH * DD * EE;
  const size_t nWp = (size_t)HH * EE * DD;
  const size_t nH  = (size_t)BB * HH * SS * EE;
  const size_t nO  = (size_t)BB * SS * HH * EE;

  char* ws = (char*)d_ws;
  size_t off = 0;
  auto carve = [&](size_t bytes) -> char* {
    char* p = ws + off;
    off += (bytes + 255) & ~(size_t)255;
    return p;
  };
  bf16_t* Xb  = (bf16_t*)carve(nX * 2);
  bf16_t* Wqb = (bf16_t*)carve(nW * 2);
  bf16_t* Wkb = (bf16_t*)carve(nW * 2);
  bf16_t* Wvb = (bf16_t*)carve(nW * 2);
  bf16_t* Wpb = (bf16_t*)carve(nWp * 2);
  bf16_t* Qb  = (bf16_t*)carve(nH * 2);
  bf16_t* Kb  = (bf16_t*)carve(nH * 2);
  bf16_t* Vb  = (bf16_t*)carve(nH * 2);
  bf16_t* Ob  = (bf16_t*)carve(nO * 2);

  auto cvt = [&](const float* s, bf16_t* d, size_t n) {
    int blocks = (int)((n + 1023) / 1024);
    if (blocks > 4096) blocks = 4096;
    cvt_bf16_k<<<blocks, 256, 0, stream>>>(s, d, (int)n);
  };
  cvt(x, Xb, nX);
  cvt(Wq, Wqb, nW);
  cvt(Wk, Wkb, nW);
  cvt(Wv, Wvb, nW);
  cvt(Wp, Wpb, nWp);

  // Q/K/V projections: M = B*S = 8192, N = E = 512, K = D = 512, per head.
  dim3 gQKV(EE / 128, (BB * SS) / 128, HH);
  gemm128<0><<<gQKV, 256, 0, stream>>>(Xb, DD, Wqb, EE, DD, bq, (void*)Qb);
  gemm128<0><<<gQKV, 256, 0, stream>>>(Xb, DD, Wkb, EE, DD, bk, (void*)Kb);
  gemm128<0><<<gQKV, 256, 0, stream>>>(Xb, DD, Wvb, EE, DD, bv, (void*)Vb);

  // Single-pass fused flash attention.
  attn_fused<<<dim3(SS / 16, BB * HH), 256, 0, stream>>>(Qb, Kb, Vb, Ob);

  // Output projection: M = B*S, N = D = 512, K = H*E = 4096.
  dim3 gOut(DD / 128, (BB * SS) / 128, 1);
  gemm128<1><<<gOut, 256, 0, stream>>>(Ob, HH * EE, Wpb, DD, HH * EE, bp, (void*)out);
}
